// ESN_84842783965686
// MI455X (gfx1250) — matrix operand order repaired and checked by reading, not compiled, not run on hardware
//
#include <hip/hip_runtime.h>

// ----------------------------------------------------------------------------
// ESN on MI455X (gfx1250, wave32):
//   U = W_in @ X^T  (f16 WMMA, f32 accum)  fused with per-row diagonal scan
//   s_i(t) = tanh(u_i(t) + d_i * s_i(t-1)),  then  out = W_out @ s(T-1)
//
// The 4096-step serial tanh recurrence (8192-way parallel -> 256 waves) is the
// latency floor (~4096 x (fma+tanh) cycles). The WMMA input-drive GEMM
// (6.4 GFLOP) and all memory traffic (~6 MB, L2-resident) must hide under it.
// B fragments are software-pipelined in registers (ping-pong, 2x unroll) so
// the L2 load-to-use latency of tile t+1 is covered by the scan of tile t.
// ----------------------------------------------------------------------------

typedef _Float16 v8h  __attribute__((ext_vector_type(8)));
typedef _Float16 v16h __attribute__((ext_vector_type(16)));
typedef float    v4f  __attribute__((ext_vector_type(4)));
typedef float    v8f  __attribute__((ext_vector_type(8)));

#define RESERVOIR    8192
#define WINDOW       96
#define PRED         96
#define SEGMENTS     4096
#define N_TILE       16                      // timesteps per WMMA tile (N dim)
#define ROWS_PER_WAVE 32                     // 2 x 16-row WMMA M-strips
#define WAVES_PER_WG 4
#define ROWS_PER_WG  (ROWS_PER_WAVE * WAVES_PER_WG)   // 128
#define NUM_WG       (RESERVOIR / ROWS_PER_WG)        // 64
#define NUM_TILES    (SEGMENTS / N_TILE)              // 256

__device__ __forceinline__ float esn_tanh(float z) {
#if __has_builtin(__builtin_amdgcn_tanhf)
  return __builtin_amdgcn_tanhf(z);          // gfx1250 v_tanh_f32
#elif __has_builtin(__builtin_amdgcn_tanh_f32)
  return __builtin_amdgcn_tanh_f32(z);
#else
  // tanh(z) = 1 - 2/(exp2(2*log2e*z)+1), via v_exp_f32 / v_rcp_f32
  float e = __builtin_amdgcn_exp2f(z * 2.8853900817779268f);
  return 1.0f - 2.0f * __builtin_amdgcn_rcpf(e + 1.0f);
#endif
}

// ---------------------------------------------------------------------------
// Kernel 1: f32 -> f16 conversion of W_in and x (both stay row-major; the
// WMMA fragment layouts below read them with contiguous 16B/32B chunks).
// ---------------------------------------------------------------------------
__global__ void esn_convert_f16(const float* __restrict__ W_in,
                                const float* __restrict__ x,
                                _Float16* __restrict__ Wh,
                                _Float16* __restrict__ Xh) {
  const int N1 = RESERVOIR * WINDOW;
  const int N2 = SEGMENTS * WINDOW;
  int i = blockIdx.x * blockDim.x + threadIdx.x;
  if (i < N1) {
    Wh[i] = (_Float16)W_in[i];
  } else {
    int j = i - N1;
    if (j < N2) Xh[j] = (_Float16)x[j];
  }
}

// ---------------------------------------------------------------------------
// Helpers for the fused kernel (forceinlined; everything stays in registers).
// ---------------------------------------------------------------------------
__device__ __forceinline__ void load_b(v16h b[3], const _Float16* __restrict__ Xh,
                                       int t0, int n16, int k16) {
  // B[k][n] = X[t0+n][k]; per lane one contiguous 32B chunk per K-block.
  const _Float16* xp = Xh + (size_t)(t0 + n16) * WINDOW + k16;
  b[0] = *(const v16h*)(xp);                 // K = 0..31  (this half-wave's 16)
  b[1] = *(const v16h*)(xp + 32);            // K = 32..63
  b[2] = *(const v16h*)(xp + 64);            // K = 64..95
  // Stage X two tiles ahead into GL2 (speculative; stays inside workspace).
  __builtin_prefetch(xp + 2 * N_TILE * WINDOW, 0, 1);
}

__device__ __forceinline__ void tile_step(const v16h a[2][3], const v16h b[3],
                                          float* __restrict__ ldsW,  // [N_TILE][32]
                                          int lane, int n16, int mofs,
                                          float dv, float& s) {
  // U-tile: 6x v_wmma_f32_16x16x32_f16 (2 M-strips x 3 K-steps), staged
  // through per-wave LDS in [n][m] order (contiguous ds_store_b128).
#pragma unroll
  for (int f = 0; f < 2; ++f) {
    v8f c = {};
    c = __builtin_amdgcn_wmma_f32_16x16x32_f16(false, a[f][0], false, b[0],
                                               (short)0, c, false, false);
    c = __builtin_amdgcn_wmma_f32_16x16x32_f16(false, a[f][1], false, b[1],
                                               (short)0, c, false, false);
    c = __builtin_amdgcn_wmma_f32_16x16x32_f16(false, a[f][2], false, b[2],
                                               (short)0, c, false, false);
    // C layout: lane holds column n16, rows m = mofs+16f .. +7 (contiguous)
    float* dst = ldsW + n16 * ROWS_PER_WAVE + mofs + 16 * f;
    *(v4f*)(dst)     = __builtin_shufflevector(c, c, 0, 1, 2, 3);
    *(v4f*)(dst + 4) = __builtin_shufflevector(c, c, 4, 5, 6, 7);
  }
  // Serial diagonal scan: lane owns row `lane`; conflict-free ds_load_b32.
  // Per-wave LDS ops are in-order, so no barrier is needed.
#pragma unroll
  for (int n = 0; n < N_TILE; ++n) {
    float u = ldsW[n * ROWS_PER_WAVE + lane];
    s = esn_tanh(__builtin_fmaf(dv, s, u));
  }
}

// ---------------------------------------------------------------------------
// Kernel 2: fused WMMA input-drive + diagonal tanh scan.
// Each wave owns 32 reservoir rows; its W_in slice lives in registers as six
// v16h A-fragments. B fragments ping-pong across a 2x-unrolled tile loop so
// global load latency hides under the previous tile's scan chain.
// ---------------------------------------------------------------------------
__global__ __launch_bounds__(ROWS_PER_WG)
void esn_scan_wmma(const _Float16* __restrict__ Wh,
                   const _Float16* __restrict__ Xh,
                   const float* __restrict__ d,
                   float* __restrict__ s_final) {
  __shared__ __align__(32) float lds[WAVES_PER_WG][N_TILE][ROWS_PER_WAVE];

  const int lane    = threadIdx.x & 31;
  const int wave    = threadIdx.x >> 5;
  const int rowBase = (blockIdx.x * WAVES_PER_WG + wave) * ROWS_PER_WAVE;
  float* ldsW = &lds[wave][0][0];

  // --- A fragments: contiguous k order (k = 16h + s), same as B ---------------------
  // lane&15 = M row within strip; lane>>4 selects K-half (+0 / +16)
  v16h a[2][3];
  {
    const int m16 = lane & 15;
    const int k8  = (lane >> 4) << 4;        // 0 or 16
#pragma unroll
    for (int f = 0; f < 2; ++f) {
      const _Float16* rp = Wh + (size_t)(rowBase + 16 * f + m16) * WINDOW;
#pragma unroll
      for (int kb = 0; kb < 3; ++kb) {       // K blocks 0,32,64
        a[f][kb] = *(const v16h*)(rp + kb * 32 + k8);     // K = kb*32 + k8 + 0..15
      }
    }
  }

  const float dv = d[rowBase + lane];        // this lane's diagonal element
  float s = 0.0f;                            // reservoir state, in-register

  const int n16  = lane & 15;                // B/C column owned by this lane
  const int k16  = (lane >> 4) << 4;         // B K-half: +0 or +16
  const int mofs = (lane >> 4) << 3;         // C rows: m = mofs..+7 (+16f)

  // --- software-pipelined tile loop (ping-pong B registers, 2x unroll) ------
  v16h bA[3], bB[3];
  load_b(bA, Xh, 0, n16, k16);
  for (int tile = 0; tile < NUM_TILES; tile += 2) {
    // issue loads for tile+1, then consume tile (scan covers the latency)
    load_b(bB, Xh, (tile + 1) * N_TILE, n16, k16);
    tile_step(a, bA, ldsW, lane, n16, mofs, dv, s);
    // issue loads for tile+2 (clamped on the last iteration), consume tile+1
    const int t2 = (tile + 2 < NUM_TILES) ? tile + 2 : tile;
    load_b(bA, Xh, t2 * N_TILE, n16, k16);
    tile_step(a, bB, ldsW, lane, n16, mofs, dv, s);
  }

  s_final[rowBase + lane] = s;
}

// ---------------------------------------------------------------------------
// Kernel 3: out = W_out @ s_final  (96 x 8192 GEMV, one block per output row)
// ---------------------------------------------------------------------------
__global__ __launch_bounds__(256)
void esn_project(const float* __restrict__ W_out,
                 const float* __restrict__ s,
                 float* __restrict__ out) {
  __shared__ float red[256];
  const int row = blockIdx.x;
  const float* wp = W_out + (size_t)row * RESERVOIR;
  float acc = 0.0f;
  for (int i = threadIdx.x; i < RESERVOIR; i += 256)
    acc = __builtin_fmaf(wp[i], s[i], acc);
  red[threadIdx.x] = acc;
  __syncthreads();
  for (int off = 128; off > 0; off >>= 1) {
    if (threadIdx.x < off) red[threadIdx.x] += red[threadIdx.x + off];
    __syncthreads();
  }
  if (threadIdx.x == 0) out[row] = red[0];
}

// ---------------------------------------------------------------------------
// Launch. Workspace layout (bytes):
//   [0)                Wh      : 8192*96*2  = 1,572,864
//   [1,572,864)        Xh      : 4096*96*2  =   786,432
//   [2,359,296)        s_final : 8192*4     =    32,768     (total ~2.3 MB)
// ---------------------------------------------------------------------------
extern "C" void kernel_launch(void* const* d_in, const int* in_sizes, int n_in,
                              void* d_out, int out_size, void* d_ws, size_t ws_size,
                              hipStream_t stream) {
  const float* x     = (const float*)d_in[0];
  const float* W_in  = (const float*)d_in[1];
  const float* d     = (const float*)d_in[2];
  const float* W_out = (const float*)d_in[3];

  char* ws = (char*)d_ws;
  _Float16* Wh      = (_Float16*)(ws);
  _Float16* Xh      = (_Float16*)(ws + (size_t)RESERVOIR * WINDOW * 2);
  float*    s_final = (float*)(ws + (size_t)RESERVOIR * WINDOW * 2
                                  + (size_t)SEGMENTS * WINDOW * 2);

  const int totalConv = RESERVOIR * WINDOW + SEGMENTS * WINDOW;
  esn_convert_f16<<<(totalConv + 255) / 256, 256, 0, stream>>>(W_in, x, Wh, Xh);
  esn_scan_wmma<<<NUM_WG, ROWS_PER_WG, 0, stream>>>(Wh, Xh, d, s_final);
  esn_project<<<PRED, 256, 0, stream>>>(W_out, s_final, (float*)d_out);
}
